// MoE_28922309771627
// MI455X (gfx1250) — matrix operand order repaired and checked by reading, not compiled, not run on hardware
//
#include <hip/hip_runtime.h>
#include <hip/hip_bf16.h>

typedef __attribute__((ext_vector_type(16))) __bf16 bf16x16;
typedef __attribute__((ext_vector_type(8)))  __bf16 bf16x8;
typedef __attribute__((ext_vector_type(4)))  __bf16 bf16x4;
typedef __attribute__((ext_vector_type(8)))  float  f32x8;
typedef int v4i_vs __attribute__((vector_size(16)));   // matches builtin param type

#define AS1 __attribute__((address_space(1)))
#define AS3 __attribute__((address_space(3)))

#if defined(__has_builtin)
#if __has_builtin(__builtin_amdgcn_global_load_async_to_lds_b128) && \
    __has_builtin(__builtin_amdgcn_s_wait_asynccnt)
#define MOE_ASYNC_LDS 1
#endif
#endif

// Fixed problem dims from setup_inputs()
constexpr int T   = 4096;   // B*S tokens
constexpr int D   = 2048;   // d_model
constexpr int E   = 8;      // routed experts
constexpr int NI  = 1024;   // moe_inter_dim
constexpr int ISH = 1024;   // shared width
constexpr int NSLOT = 2 * T;

constexpr int TM = 64;          // rows per block tile
constexpr int LROW = 40;        // LDS row stride in bf16 elems (80B, conflict-free)

__device__ __forceinline__ int imin(int a, int b) { return a < b ? a : b; }

// ---------------- workspace layout (bytes) ----------------
constexpr size_t ALN = 256;
constexpr size_t al(size_t x) { return (x + ALN - 1) & ~(ALN - 1); }
constexpr size_t OFF_XB   = 0;
constexpr size_t OFF_W1B  = al(OFF_XB   + (size_t)T * D * 2);
constexpr size_t OFF_W3B  = al(OFF_W1B  + (size_t)E * NI * D * 2);
constexpr size_t OFF_W2B  = al(OFF_W3B  + (size_t)E * NI * D * 2);
constexpr size_t OFF_WS1B = al(OFF_W2B  + (size_t)E * D * NI * 2);
constexpr size_t OFF_WS3B = al(OFF_WS1B + (size_t)ISH * D * 2);
constexpr size_t OFF_WS2B = al(OFF_WS3B + (size_t)ISH * D * 2);
constexpr size_t OFF_HB   = al(OFF_WS2B + (size_t)D * ISH * 2);
constexpr size_t OFF_HS   = al(OFF_HB   + (size_t)NSLOT * NI * 2);
constexpr size_t OFF_YB   = al(OFF_HS   + (size_t)T * ISH * 2);
constexpr size_t OFF_CNT  = al(OFF_YB   + (size_t)NSLOT * D * 4);
constexpr size_t OFF_OFFS = OFF_CNT  + 64;
constexpr size_t OFF_CUR  = OFF_OFFS + 64;
constexpr size_t OFF_LIST = al(OFF_CUR + 64);
constexpr size_t OFF_LWT  = al(OFF_LIST + (size_t)NSLOT * 4);
constexpr size_t OFF_POS  = al(OFF_LWT  + (size_t)NSLOT * 4);
constexpr size_t OFF_TKE  = al(OFF_POS  + (size_t)NSLOT * 4);
constexpr size_t OFF_TKW  = al(OFF_TKE  + (size_t)NSLOT * 4);

// ---------------- async/LDS staging helpers ----------------
__device__ __forceinline__ void stage_b128(const __bf16* g, __bf16* l) {
#if MOE_ASYNC_LDS
  __builtin_amdgcn_global_load_async_to_lds_b128(
      (AS1 v4i_vs*)g, (AS3 v4i_vs*)l, 0, 0);
#else
  *(bf16x8*)l = *(const bf16x8*)g;
#endif
}
__device__ __forceinline__ void stage_wait(int n) {
#if MOE_ASYNC_LDS
  if (n == 0) __builtin_amdgcn_s_wait_asynccnt(0);
  else        __builtin_amdgcn_s_wait_asynccnt(2);
#else
  (void)n;
#endif
}

// ---------------- fp32 -> bf16 streaming convert ----------------
__global__ __launch_bounds__(256) void moe_cvt_bf16(const float* __restrict__ src,
                                                    __bf16* __restrict__ dst, int n4) {
  int i = blockIdx.x * blockDim.x + threadIdx.x;
  if (i >= n4) return;
  float4 v = ((const float4*)src)[i];
  bf16x4 r = { (__bf16)v.x, (__bf16)v.y, (__bf16)v.z, (__bf16)v.w };
  ((bf16x4*)dst)[i] = r;
}

__global__ void moe_init_meta(int* counts) {
  if (threadIdx.x < E) counts[threadIdx.x] = 0;
}

// ---------------- gating: one wave per token ----------------
__global__ __launch_bounds__(256) void moe_gate(const float* __restrict__ x,
                                                const float* __restrict__ gw,
                                                const float* __restrict__ gb,
                                                int* __restrict__ tk_e,
                                                float* __restrict__ tk_w,
                                                int* __restrict__ counts) {
  int t = blockIdx.x * 8 + (threadIdx.x >> 5);
  int lane = threadIdx.x & 31;
  if (t >= T) return;
  const float* xr = x + (size_t)t * D;
  float acc[E];
#pragma unroll
  for (int e = 0; e < E; ++e) acc[e] = 0.0f;
  for (int k = lane; k < D; k += 32) {
    float xv = xr[k];
#pragma unroll
    for (int e = 0; e < E; ++e) acc[e] += xv * gw[(size_t)e * D + k];
  }
#pragma unroll
  for (int e = 0; e < E; ++e)
#pragma unroll
    for (int off = 16; off; off >>= 1) acc[e] += __shfl_xor(acc[e], off, 32);
  if (lane == 0) {
    float mx = acc[0];
#pragma unroll
    for (int e = 1; e < E; ++e) mx = fmaxf(mx, acc[e]);
    float s[E], se = 0.0f;
#pragma unroll
    for (int e = 0; e < E; ++e) { s[e] = __expf(acc[e] - mx); se += s[e]; }
    float inv = 1.0f / se;
#pragma unroll
    for (int e = 0; e < E; ++e) s[e] *= inv;
    int i0 = 0, i1 = -1;
    float b0 = -1e30f, b1 = -1e30f;
#pragma unroll
    for (int e = 0; e < E; ++e) {
      float bv = s[e] + gb[e];
      if (bv > b0) { b1 = b0; i1 = i0; b0 = bv; i0 = e; }
      else if (bv > b1) { b1 = bv; i1 = e; }
    }
    tk_e[t * 2 + 0] = i0; tk_w[t * 2 + 0] = s[i0];
    tk_e[t * 2 + 1] = i1; tk_w[t * 2 + 1] = s[i1];
    atomicAdd(&counts[i0], 1);
    atomicAdd(&counts[i1], 1);
  }
}

__global__ void moe_scan(const int* __restrict__ counts, int* __restrict__ offs,
                         int* __restrict__ cursors) {
  if (threadIdx.x == 0 && blockIdx.x == 0) {
    int s = 0;
    for (int e = 0; e < E; ++e) { offs[e] = s; cursors[e] = s; s += counts[e]; }
  }
}

__global__ __launch_bounds__(256) void moe_scatter(const int* __restrict__ tk_e,
                                                   const float* __restrict__ tk_w,
                                                   int* __restrict__ cursors,
                                                   int* __restrict__ tok_list,
                                                   float* __restrict__ tok_wt,
                                                   int* __restrict__ pos) {
  int i = blockIdx.x * blockDim.x + threadIdx.x;
  if (i >= NSLOT) return;
  int e = tk_e[i];
  int p = atomicAdd(&cursors[e], 1);
  tok_list[p] = i >> 1;
  tok_wt[p] = tk_w[i];
  pos[i] = p;
}

// ============ up-projection: h = silu(x W1^T) * (x W3^T) ============
// Tile: 64 rows x 64 cols per 128-thread block; A staged via async->LDS,
// double-buffered; each wave does 8 WMMAs per k-step (2 mats x 4 M-subtiles).
__global__ __launch_bounds__(128) void moe_upproj(
    const __bf16* __restrict__ xb, const int* __restrict__ tok_list,
    const int* __restrict__ segOff, const int* __restrict__ segCnt,
    const __bf16* __restrict__ Wg, const __bf16* __restrict__ Wu,
    __bf16* __restrict__ hOut, int Kdim, int Ndim, int tilesPerExpert) {
  __shared__ __bf16 smem[2][TM * LROW];   // 2 x 5120B

  int e   = blockIdx.x / tilesPerExpert;
  int til = blockIdx.x % tilesPerExpert;
  int segStart = segOff ? segOff[e] : 0;
  int cnt      = segCnt ? segCnt[e] : tilesPerExpert * TM;
  int row0 = til * TM;
  if (row0 >= cnt) return;

  int tid  = threadIdx.x;
  int lane = tid & 31;
  int wave = tid >> 5;
  int half = lane >> 4;                 // 0 or 1
  int m    = lane & 15;
  int n    = blockIdx.y * 64 + wave * 16 + (lane & 15);

  // ---- A staging addresses: 2 threads per row, 16 elems (32B) each ----
  int srow   = tid >> 1;
  int schunk = (tid & 1) * 16;
  int srcl   = imin(row0 + srow, cnt - 1);
  int stok   = tok_list ? tok_list[segStart + srcl] : srcl;
  const __bf16* sg = xb + (size_t)stok * Kdim + schunk;
  __bf16* sl[2] = { &smem[0][srow * LROW + schunk], &smem[1][srow * LROW + schunk] };

  // ---- B row pointers (per wave N column) ----
  const __bf16* w1row = Wg + ((size_t)e * Ndim + n) * Kdim;
  const __bf16* w3row = Wu + ((size_t)e * Ndim + n) * Kdim;
  int kB = half * 16;

  f32x8 acc1[4], acc3[4];
#pragma unroll
  for (int mt = 0; mt < 4; ++mt) {
    acc1[mt] = (f32x8){0,0,0,0,0,0,0,0};
    acc3[mt] = (f32x8){0,0,0,0,0,0,0,0};
  }

  // prologue: stage k=0 into buffer 0
  stage_b128(sg, sl[0]);
  stage_b128(sg + 8, sl[0] + 8);

  for (int k = 0; k < Kdim; k += 32) {
    int cur = (k >> 5) & 1;
    if (k + 32 < Kdim) {
      stage_b128(sg + k + 32, sl[cur ^ 1]);
      stage_b128(sg + k + 40, sl[cur ^ 1] + 8);
      stage_wait(2);
    } else {
      stage_wait(0);
    }
    __syncthreads();

    __builtin_prefetch(w1row + k + 256, 0, 1);
    __builtin_prefetch(w3row + k + 256, 0, 1);
    bf16x16 b1 = *(const bf16x16*)(w1row + k + kB);
    bf16x16 b3 = *(const bf16x16*)(w3row + k + kB);

    const __bf16* abase = &smem[cur][0];
#pragma unroll
    for (int mt = 0; mt < 4; ++mt) {
      const __bf16* ar = abase + (size_t)(mt * 16 + m) * LROW + half * 16;
      bf16x8 alo = *(const bf16x8*)ar;          // K = kbase..kbase+7
      bf16x8 ahi = *(const bf16x8*)(ar + 8);    // K = kbase+8..kbase+15
      bf16x16 a = __builtin_shufflevector(alo, ahi, 0,1,2,3,4,5,6,7,8,9,10,11,12,13,14,15);
      acc1[mt] = __builtin_amdgcn_wmma_f32_16x16x32_bf16(false, a, false, b1, (short)0, acc1[mt], false, false);
      acc3[mt] = __builtin_amdgcn_wmma_f32_16x16x32_bf16(false, a, false, b3, (short)0, acc3[mt], false, false);
    }
    __syncthreads();
  }

  int mhi = half * 8;   // C layout: lane<16 -> M=r, lane>=16 -> M=r+8
#pragma unroll
  for (int mt = 0; mt < 4; ++mt) {
#pragma unroll
    for (int r = 0; r < 8; ++r) {
      int rr = row0 + mt * 16 + r + mhi;
      if (rr < cnt) {
        float g = acc1[mt][r];
        float u = acc3[mt][r];
        float h = (g / (1.0f + __expf(-g))) * u;    // silu(g) * u
        hOut[(size_t)(segStart + rr) * Ndim + n] = (__bf16)h;
      }
    }
  }
}

// ============ down-projection: out = (h W2^T) [* wt] ============
__global__ __launch_bounds__(128) void moe_downproj(
    const __bf16* __restrict__ A, const __bf16* __restrict__ W,
    float* __restrict__ Out, const int* __restrict__ segOff,
    const int* __restrict__ segCnt, const float* __restrict__ wt,
    int Kdim, int Ndim, int tilesPerExpert) {
  __shared__ __bf16 smem[2][TM * LROW];

  int e   = blockIdx.x / tilesPerExpert;
  int til = blockIdx.x % tilesPerExpert;
  int segStart = segOff ? segOff[e] : 0;
  int cnt      = segCnt ? segCnt[e] : tilesPerExpert * TM;
  int row0 = til * TM;
  if (row0 >= cnt) return;

  int tid  = threadIdx.x;
  int lane = tid & 31;
  int wave = tid >> 5;
  int half = lane >> 4;
  int m    = lane & 15;
  int n    = blockIdx.y * 64 + wave * 16 + (lane & 15);

  int srow   = tid >> 1;
  int schunk = (tid & 1) * 16;
  int srcl   = imin(row0 + srow, cnt - 1);
  const __bf16* sg = A + (size_t)(segStart + srcl) * Kdim + schunk;
  __bf16* sl[2] = { &smem[0][srow * LROW + schunk], &smem[1][srow * LROW + schunk] };

  const __bf16* wrow = W + ((size_t)e * Ndim + n) * Kdim;
  int kB = half * 16;

  f32x8 acc[4];
#pragma unroll
  for (int mt = 0; mt < 4; ++mt) acc[mt] = (f32x8){0,0,0,0,0,0,0,0};

  stage_b128(sg, sl[0]);
  stage_b128(sg + 8, sl[0] + 8);

  for (int k = 0; k < Kdim; k += 32) {
    int cur = (k >> 5) & 1;
    if (k + 32 < Kdim) {
      stage_b128(sg + k + 32, sl[cur ^ 1]);
      stage_b128(sg + k + 40, sl[cur ^ 1] + 8);
      stage_wait(2);
    } else {
      stage_wait(0);
    }
    __syncthreads();

    __builtin_prefetch(wrow + k + 256, 0, 1);
    bf16x16 b = *(const bf16x16*)(wrow + k + kB);

    const __bf16* abase = &smem[cur][0];
#pragma unroll
    for (int mt = 0; mt < 4; ++mt) {
      const __bf16* ar = abase + (size_t)(mt * 16 + m) * LROW + half * 16;
      bf16x8 alo = *(const bf16x8*)ar;
      bf16x8 ahi = *(const bf16x8*)(ar + 8);
      bf16x16 a = __builtin_shufflevector(alo, ahi, 0,1,2,3,4,5,6,7,8,9,10,11,12,13,14,15);
      acc[mt] = __builtin_amdgcn_wmma_f32_16x16x32_bf16(false, a, false, b, (short)0, acc[mt], false, false);
    }
    __syncthreads();
  }

  int mhi = half * 8;
#pragma unroll
  for (int mt = 0; mt < 4; ++mt) {
#pragma unroll
    for (int r = 0; r < 8; ++r) {
      int rr = row0 + mt * 16 + r + mhi;
      if (rr < cnt) {
        float s = wt ? wt[segStart + rr] : 1.0f;
        Out[(size_t)(segStart + rr) * Ndim + n] = acc[mt][r] * s;
      }
    }
  }
}

// ------- combine: out += yb[slot0] + yb[slot1] (fixed order, deterministic) ---
__global__ __launch_bounds__(256) void moe_combine(float* __restrict__ out,
                                                   const float* __restrict__ yb,
                                                   const int* __restrict__ pos) {
  int i = blockIdx.x * blockDim.x + threadIdx.x;   // over T*D/4
  if (i >= T * (D / 4)) return;
  int t  = i >> 9;                                 // D/4 == 512
  int d4 = i & 511;
  int p0 = pos[t * 2 + 0];
  int p1 = pos[t * 2 + 1];
  float4 o  = ((float4*)out)[i];
  float4 y0 = ((const float4*)(yb + (size_t)p0 * D))[d4];
  float4 y1 = ((const float4*)(yb + (size_t)p1 * D))[d4];
  o.x += y0.x + y1.x; o.y += y0.y + y1.y; o.z += y0.z + y1.z; o.w += y0.w + y1.w;
  ((float4*)out)[i] = o;
}

extern "C" void kernel_launch(void* const* d_in, const int* in_sizes, int n_in,
                              void* d_out, int out_size, void* d_ws, size_t ws_size,
                              hipStream_t stream) {
  const float* x   = (const float*)d_in[0];
  const float* gw  = (const float*)d_in[1];
  const float* gb  = (const float*)d_in[2];
  const float* W1  = (const float*)d_in[3];
  const float* W2  = (const float*)d_in[4];
  const float* W3  = (const float*)d_in[5];
  const float* Ws1 = (const float*)d_in[6];
  const float* Ws2 = (const float*)d_in[7];
  const float* Ws3 = (const float*)d_in[8];
  float* out = (float*)d_out;

  char* w = (char*)d_ws;
  __bf16* xb   = (__bf16*)(w + OFF_XB);
  __bf16* W1b  = (__bf16*)(w + OFF_W1B);
  __bf16* W3b  = (__bf16*)(w + OFF_W3B);
  __bf16* W2b  = (__bf16*)(w + OFF_W2B);
  __bf16* Ws1b = (__bf16*)(w + OFF_WS1B);
  __bf16* Ws3b = (__bf16*)(w + OFF_WS3B);
  __bf16* Ws2b = (__bf16*)(w + OFF_WS2B);
  __bf16* hbuf = (__bf16*)(w + OFF_HB);
  __bf16* hs   = (__bf16*)(w + OFF_HS);
  float*  ybuf = (float*)(w + OFF_YB);
  int* counts  = (int*)(w + OFF_CNT);
  int* offs    = (int*)(w + OFF_OFFS);
  int* cursors = (int*)(w + OFF_CUR);
  int* tok_list= (int*)(w + OFF_LIST);
  float* tok_wt= (float*)(w + OFF_LWT);
  int* pos     = (int*)(w + OFF_POS);
  int* tk_e    = (int*)(w + OFF_TKE);
  float* tk_w  = (float*)(w + OFF_TKW);

  // 1) fp32 -> bf16 conversions
  auto cvt = [&](const float* s, __bf16* d, size_t n) {
    int n4 = (int)(n / 4);
    moe_cvt_bf16<<<(n4 + 255) / 256, 256, 0, stream>>>(s, d, n4);
  };
  cvt(x,   xb,   (size_t)T * D);
  cvt(W1,  W1b,  (size_t)E * NI * D);
  cvt(W3,  W3b,  (size_t)E * NI * D);
  cvt(W2,  W2b,  (size_t)E * D * NI);
  cvt(Ws1, Ws1b, (size_t)ISH * D);
  cvt(Ws3, Ws3b, (size_t)ISH * D);
  cvt(Ws2, Ws2b, (size_t)D * ISH);

  // 2) gating + dispatch lists
  moe_init_meta<<<1, 32, 0, stream>>>(counts);
  moe_gate<<<T / 8, 256, 0, stream>>>(x, gw, gb, tk_e, tk_w, counts);
  moe_scan<<<1, 1, 0, stream>>>(counts, offs, cursors);
  moe_scatter<<<NSLOT / 256, 256, 0, stream>>>(tk_e, tk_w, cursors, tok_list, tok_wt, pos);

  // 3) shared expert: hs = silu(x Ws1^T)*(x Ws3^T); out = hs Ws2^T
  moe_upproj<<<dim3(T / TM, ISH / 64), 128, 0, stream>>>(
      xb, nullptr, nullptr, nullptr, Ws1b, Ws3b, hs, D, ISH, T / TM);
  moe_downproj<<<dim3(T / TM, D / 64), 128, 0, stream>>>(
      hs, Ws2b, out, nullptr, nullptr, nullptr, ISH, D, T / TM);

  // 4) routed experts: hbuf = silu(x W1^T)*(x W3^T); ybuf = wt * (hbuf W2^T)
  moe_upproj<<<dim3(E * (T / TM), NI / 64), 128, 0, stream>>>(
      xb, tok_list, offs, counts, W1b, W3b, hbuf, D, NI, T / TM);
  moe_downproj<<<dim3(E * (T / TM), D / 64), 128, 0, stream>>>(
      hbuf, W2b, ybuf, offs, counts, tok_wt, NI, D, T / TM);

  // 5) deterministic combine
  moe_combine<<<(T * (D / 4) + 255) / 256, 256, 0, stream>>>(out, ybuf, pos);
}
